// SetGNN_20358144983693
// MI455X (gfx1250) — matrix operand order repaired and checked by reading, not compiled, not run on hardware
//
#include <hip/hip_runtime.h>
#include <hip/hip_bf16.h>
#include <stdint.h>

// ---------------------------------------------------------------------------
// SetGNN (AllSet) forward for MI455X / gfx1250.
//  - All 18 linear layers run through v_wmma_f32_16x16x32_bf16 (f32 accum).
//  - Fused per-block MLP kernel: LN0 -> GEMM1 -> bias+relu -> LN1 -> GEMM2,
//    RT row-tiles (16 rows each) per block so every B (weight) fragment is
//    reused RT times -> 4x less L2 weight traffic on the big passes.
//  - Scatter-mean's division is folded away: LN is scale-invariant per row and
//    the consumer MLP always begins with LN0, so segment_sum alone suffices.
//  - Scatter uses f32 global atomics, agent scope, relaxed ordering.
// ---------------------------------------------------------------------------

typedef __attribute__((ext_vector_type(16))) __bf16 v16bf;
typedef __attribute__((ext_vector_type(8)))  __bf16 v8bf;
typedef __attribute__((ext_vector_type(8)))  float  v8f;

static __device__ __forceinline__ __bf16 f2bf(float f) {
  unsigned u = __builtin_bit_cast(unsigned, f);
  unsigned r = u + 0x7FFFu + ((u >> 16) & 1u);   // round-to-nearest-even
  return __builtin_bit_cast(__bf16, (unsigned short)(r >> 16));
}

// A fragment: 16x32 bf16 tile, row-major source (leading dim ldk), per ISA
// 7.12.2: lanes 0-15 hold row m, K = k0+[0..7] and k0+[16..23];
//         lanes 16-31 hold row m, K = k0+[8..15] and k0+[24..31].
static __device__ __forceinline__ v16bf load_a_frag(const __bf16* base, int ldk,
                                                    int lane, int k0) {
  const int m    = lane & 15;
  const int koff = (lane & 16) ? 8 : 0;
  const __bf16* p = base + m * ldk + k0 + koff;
  union { v16bf v; v8bf h[2]; } u;
  u.h[0] = *(const v8bf*)(p);
  u.h[1] = *(const v8bf*)(p + 16);
  return u.v;
}

// B fragment: 32x16 bf16 from *transposed* weights WT[n][k] (leading dim ldk).
// Lanes 0-15: col n, K = k0+[0..7] and k0+[16..23]; lanes 16-31: col n, K = k0+[8..15] and k0+[24..31].
static __device__ __forceinline__ v16bf load_b_frag(const __bf16* wt_n0, int ldk,
                                                    int lane, int k0) {
  const int n    = lane & 15;
  const int koff = (lane & 16) ? 8 : 0;
  const __bf16* p = wt_n0 + (size_t)n * ldk + k0 + koff;
  union { v16bf v; v8bf h[2]; } u;
  u.h[0] = *(const v8bf*)(p);
  u.h[1] = *(const v8bf*)(p + 16);
  return u.v;
}

struct MlpP {
  const float *b0, *b1, *ln0_b, *ln0_g, *ln1_b, *ln1_g;
};

// Fused MLP: y = [relu]( LN1( relu( [LN0](x) @ W0 + b0 ) ) @ W1 + b1 )
// One block = RT*16 rows, 4 waves; wave w owns output tiles n = w, w+4, ...;
// each B fragment is shared by the RT row-tiles.
template<int DIN, int DHID, int DOUTP, int DOUT, bool HAS_LN0, bool RELU_OUT, int RT>
__global__ __launch_bounds__(128)
void mlp_fused(const float* __restrict__ x, int nrows, MlpP p,
               const __bf16* __restrict__ Wt0, const __bf16* __restrict__ Wt1,
               float* __restrict__ out)
{
  __shared__ __align__(32) __bf16 sA [RT * 16 * DIN];
  __shared__ __align__(32) float  sH [RT * 16 * DHID];
  __shared__ __align__(32) __bf16 sA2[RT * 16 * DHID];

  const int tid   = threadIdx.x;
  const int lane  = tid & 31;
  const int wave  = tid >> 5;
  const int rowL  = tid >> 3;     // 0..15 (8 lanes cooperate per row)
  const int j     = tid & 7;
  const int rowG0 = blockIdx.x * (RT * 16);

  // ---- stage 1: optional LayerNorm over DIN -> bf16 tiles in LDS ----
  for (int rt = 0; rt < RT; ++rt) {
    const int rg = rowG0 + rt * 16 + rowL;
    const int rc = rg < nrows ? rg : (nrows - 1);
    const float* xr = x + (size_t)rc * DIN;
    __bf16* sAr = sA + rt * 16 * DIN;
    constexpr int PER = DIN / 8;
    float v[PER];
    float s = 0.f, ss = 0.f;
#pragma unroll
    for (int i = 0; i < PER; ++i) { float t = xr[j + 8 * i]; v[i] = t; s += t; ss += t * t; }
    if (HAS_LN0) {
      s += __shfl_xor(s, 1); ss += __shfl_xor(ss, 1);
      s += __shfl_xor(s, 2); ss += __shfl_xor(ss, 2);
      s += __shfl_xor(s, 4); ss += __shfl_xor(ss, 4);
      const float m   = s * (1.0f / DIN);
      const float var = ss * (1.0f / DIN) - m * m;
      const float inv = rsqrtf(var + 1e-5f);
#pragma unroll
      for (int i = 0; i < PER; ++i) {
        const int k = j + 8 * i;
        sAr[rowL * DIN + k] = f2bf((v[i] - m) * inv * p.ln0_g[k] + p.ln0_b[k]);
      }
    } else {
#pragma unroll
      for (int i = 0; i < PER; ++i) sAr[rowL * DIN + j + 8 * i] = f2bf(v[i]);
    }
  }
  __syncthreads();

  { // ---- GEMM1 (WMMA) + bias + relu -> sH (f32) ----
    constexpr int NT1  = DHID / 16;
    constexpr int TPW1 = NT1 / 4;
    v8f acc[RT][TPW1] = {};
    for (int k0 = 0; k0 < DIN; k0 += 32) {
      v16bf a[RT];
#pragma unroll
      for (int rt = 0; rt < RT; ++rt)
        a[rt] = load_a_frag(sA + rt * 16 * DIN, DIN, lane, k0);
#pragma unroll
      for (int t = 0; t < TPW1; ++t) {
        const int n0 = (wave + 4 * t) * 16;
        const v16bf b = load_b_frag(Wt0 + (size_t)n0 * DIN, DIN, lane, k0);
#pragma unroll
        for (int rt = 0; rt < RT; ++rt)
          acc[rt][t] = __builtin_amdgcn_wmma_f32_16x16x32_bf16(
              false, a[rt], false, b, (short)0, acc[rt][t], false, false);
      }
    }
    const int n16   = lane & 15;
    const int mbase = (lane & 16) ? 8 : 0;
#pragma unroll
    for (int t = 0; t < TPW1; ++t) {
      const int n = (wave + 4 * t) * 16 + n16;
      const float bias = p.b0[n];
#pragma unroll
      for (int rt = 0; rt < RT; ++rt) {
        float* sHr = sH + rt * 16 * DHID;
#pragma unroll
        for (int r = 0; r < 8; ++r)
          sHr[(mbase + r) * DHID + n] = fmaxf(acc[rt][t][r] + bias, 0.0f);
      }
    }
  }
  __syncthreads();

  // ---- LN1 over DHID -> bf16 tiles sA2 ----
  for (int rt = 0; rt < RT; ++rt) {
    constexpr int PER = DHID / 8;
    const float* hr = sH + rt * 16 * DHID + rowL * DHID;
    __bf16* sA2r = sA2 + rt * 16 * DHID;
    float v[PER];
    float s = 0.f, ss = 0.f;
#pragma unroll
    for (int i = 0; i < PER; ++i) { float t = hr[j + 8 * i]; v[i] = t; s += t; ss += t * t; }
    s += __shfl_xor(s, 1); ss += __shfl_xor(ss, 1);
    s += __shfl_xor(s, 2); ss += __shfl_xor(ss, 2);
    s += __shfl_xor(s, 4); ss += __shfl_xor(ss, 4);
    const float m   = s * (1.0f / DHID);
    const float var = ss * (1.0f / DHID) - m * m;
    const float inv = rsqrtf(var + 1e-5f);
#pragma unroll
    for (int i = 0; i < PER; ++i) {
      const int k = j + 8 * i;
      sA2r[rowL * DHID + k] = f2bf((v[i] - m) * inv * p.ln1_g[k] + p.ln1_b[k]);
    }
  }
  __syncthreads();

  { // ---- GEMM2 (WMMA) + bias (+relu) -> global ----
    constexpr int NT2  = DOUTP / 16;
    constexpr int TPW2 = (NT2 + 3) / 4;
    v8f acc[RT][TPW2] = {};
    for (int k0 = 0; k0 < DHID; k0 += 32) {
      v16bf a[RT];
#pragma unroll
      for (int rt = 0; rt < RT; ++rt)
        a[rt] = load_a_frag(sA2 + rt * 16 * DHID, DHID, lane, k0);
#pragma unroll
      for (int t = 0; t < TPW2; ++t) {
        const int nt = wave + 4 * t;
        if (nt < NT2) {
          const v16bf b = load_b_frag(Wt1 + (size_t)nt * 16 * DHID, DHID, lane, k0);
#pragma unroll
          for (int rt = 0; rt < RT; ++rt)
            acc[rt][t] = __builtin_amdgcn_wmma_f32_16x16x32_bf16(
                false, a[rt], false, b, (short)0, acc[rt][t], false, false);
        }
      }
    }
    const int n16   = lane & 15;
    const int mbase = (lane & 16) ? 8 : 0;
#pragma unroll
    for (int t = 0; t < TPW2; ++t) {
      const int nt = wave + 4 * t;
      if (nt >= NT2) break;
      const int n = nt * 16 + n16;
      if (n < DOUT) {
        const float bias = p.b1[n];
#pragma unroll
        for (int rt = 0; rt < RT; ++rt) {
#pragma unroll
          for (int r = 0; r < 8; ++r) {
            const int rg = rowG0 + rt * 16 + mbase + r;
            if (rg < nrows) {
              float vv = acc[rt][t][r] + bias;
              if (RELU_OUT) vv = fmaxf(vv, 0.0f);
              out[(size_t)rg * DOUT + n] = vv;
            }
          }
        }
      }
    }
  }
}

// Convert f32 W[din][dout] (row-major) -> bf16 WT[doutp][din] (transposed,
// zero-padded rows for dout not multiple of 16).
__global__ void cvt_wT(const float* __restrict__ W, __bf16* __restrict__ Wt,
                       int din, int dout, int doutp)
{
  const int idx = blockIdx.x * blockDim.x + threadIdx.x;
  if (idx >= doutp * din) return;
  const int n = idx / din;
  const int k = idx - n * din;
  const float v = (n < dout) ? W[(size_t)k * dout + n] : 0.0f;
  Wt[idx] = f2bf(v);
}

__global__ void fillz_kernel(float* __restrict__ p, size_t n) {
  const size_t i = (size_t)blockIdx.x * blockDim.x + threadIdx.x;
  if (i < n) p[i] = 0.0f;
}

__global__ void seti_kernel(int* p, int v) { *p = v; }

__global__ void minred_kernel(const int* __restrict__ v, int n, int* __restrict__ out) {
  __shared__ int sm;
  const int i = blockIdx.x * blockDim.x + threadIdx.x;
  if (threadIdx.x == 0) sm = 0x7fffffff;
  __syncthreads();
  const int m = (i < n) ? v[i] : 0x7fffffff;
  atomicMin(&sm, m);
  __syncthreads();
  if (threadIdx.x == 0) atomicMin(out, sm);
}

// One wave per incidence: gather 128-f32 row, scale by norm, atomically add
// into the 128-f32 target row.  Count/divide pass is intentionally omitted
// (LN scale invariance of the consumer MLP's input LayerNorm).
__global__ __launch_bounds__(256)
void scatter_kernel(const float* __restrict__ feat,
                    const int* __restrict__ gidx, const int* __restrict__ sidx,
                    const float* __restrict__ norm, const int* __restrict__ minv,
                    int gshift, int sshift, float* __restrict__ acc, int nnz)
{
  const int e = (int)((blockIdx.x * 256u + threadIdx.x) >> 5);
  if (e >= nnz) return;
  const int lane = threadIdx.x & 31;
  const int mh = *minv;
  const int g = gidx[e] - (gshift ? mh : 0);
  const int d = sidx[e] - (sshift ? mh : 0);
  const float w = norm[e];
  if (e + 64 < nnz)
    __builtin_prefetch(feat + (size_t)(gidx[e + 64] - (gshift ? mh : 0)) * 128, 0, 0);
  const float4 v = *(const float4*)(feat + (size_t)g * 128 + lane * 4);
  float* a = acc + (size_t)d * 128 + lane * 4;
  __hip_atomic_fetch_add(a + 0, w * v.x, __ATOMIC_RELAXED, __HIP_MEMORY_SCOPE_AGENT);
  __hip_atomic_fetch_add(a + 1, w * v.y, __ATOMIC_RELAXED, __HIP_MEMORY_SCOPE_AGENT);
  __hip_atomic_fetch_add(a + 2, w * v.z, __ATOMIC_RELAXED, __HIP_MEMORY_SCOPE_AGENT);
  __hip_atomic_fetch_add(a + 3, w * v.w, __ATOMIC_RELAXED, __HIP_MEMORY_SCOPE_AGENT);
}

extern "C" void kernel_launch(void* const* d_in, const int* in_sizes, int n_in,
                              void* d_out, int out_size, void* d_ws, size_t ws_size,
                              hipStream_t stream)
{
  (void)out_size; (void)ws_size;
  const float* x    = (const float*)d_in[0];
  const float* norm = (const float*)d_in[1];
  const int NNZ = in_sizes[1];
  const int NN  = in_sizes[0] / 256;     // 50000
  const int NHE = 10000;
  const int* edge_index = (const int*)d_in[n_in - 2];
  const int* src = edge_index;
  const int* he  = edge_index + NNZ;

  auto F = [&](int i) { return (const float*)d_in[i]; };
  const bool sorted = (in_sizes[2] == 256);  // jax tree-flatten (sorted keys): clf.b0 first

  // Logical MLP order: 0..3 = L1 {v2e.enc, v2e.dec, e2v.enc, e2v.dec},
  //                    4..7 = L2 same, 8 = clf.
  MlpP P[9]; const float* W0p[9]; const float* W1p[9];
  if (sorted) {
    const int base8[8] = {32, 24, 16, 8, 64, 56, 48, 40};
    for (int i = 0; i < 8; ++i) {
      const int b = base8[i];
      P[i]   = { F(b + 0), F(b + 1), F(b + 2), F(b + 3), F(b + 4), F(b + 5) };
      W0p[i] = F(b + 6); W1p[i] = F(b + 7);
    }
    P[8]   = { F(2), F(3), nullptr, nullptr, F(4), F(5) };
    W0p[8] = F(6); W1p[8] = F(7);
  } else {
    // insertion order: convs[0]{v2e{enc,dec},e2v{enc,dec}}, convs[1]..., clf
    const int base8[8] = {2, 10, 18, 26, 34, 42, 50, 58};
    for (int i = 0; i < 8; ++i) {
      const int b = base8[i];   // w0,b0,ln1_g,ln1_b,w1,b1,ln0_g,ln0_b
      P[i]   = { F(b + 1), F(b + 5), F(b + 7), F(b + 6), F(b + 3), F(b + 2) };
      W0p[i] = F(b + 0); W1p[i] = F(b + 4);
    }
    const int b = 66;
    P[8]   = { F(b + 1), F(b + 5), nullptr, nullptr, F(b + 3), F(b + 2) };
    W0p[8] = F(b + 0); W1p[8] = F(b + 4);
  }

  // ---- workspace carve ----
  char* wsb = (char*)d_ws;
  size_t off = 0;
  auto carve = [&](size_t bytes) -> void* {
    void* r = wsb + off;
    off = (off + bytes + 255) & ~(size_t)255;
    return r;
  };
  float* nodeA = (float*)carve((size_t)NN  * 128 * 4);
  float* nodeB = (float*)carve((size_t)NN  * 128 * 4);
  float* heA   = (float*)carve((size_t)NHE * 128 * 4);
  float* heB   = (float*)carve((size_t)NHE * 128 * 4);
  int*   minhe = (int*)carve(4);

  int din0[9], dh[9], dp1[9], do1[9];
  for (int i = 0; i < 9; ++i) { din0[i] = 128; dh[i] = 128; dp1[i] = 128; do1[i] = 128; }
  din0[0] = 256;                                 // L1 v2e.enc: F_IN=256
  dh[8] = 256; dp1[8] = 48; do1[8] = 40;         // clf: 128 -> 256 -> 40 (pad 48)
  __bf16* Wt0[9]; __bf16* Wt1[9];
  for (int i = 0; i < 9; ++i) {
    Wt0[i] = (__bf16*)carve((size_t)dh[i]  * din0[i] * 2);
    Wt1[i] = (__bf16*)carve((size_t)dp1[i] * dh[i]   * 2);
  }

  // ---- weight conversion (bf16, transposed, padded) ----
  for (int i = 0; i < 9; ++i) {
    int tot0 = dh[i] * din0[i];
    cvt_wT<<<(tot0 + 255) / 256, 256, 0, stream>>>(W0p[i], Wt0[i], din0[i], dh[i], dh[i]);
    int tot1 = dp1[i] * dh[i];
    cvt_wT<<<(tot1 + 255) / 256, 256, 0, stream>>>(W1p[i], Wt1[i], dh[i], do1[i], dp1[i]);
  }

  // ---- min(he) for the cidx shift ----
  seti_kernel<<<1, 1, 0, stream>>>(minhe, 0x7fffffff);
  minred_kernel<<<(NNZ + 255) / 256, 256, 0, stream>>>(he, NNZ, minhe);

  const int gN4 = (NN  + 63) / 64;   // RT=4 node passes
  const int gE4 = (NHE + 63) / 64;   // RT=4 hyperedge passes
  const int gN2 = (NN  + 31) / 32;   // RT=2 classifier
  const int sG  = (NNZ * 32 + 255) / 256;
  auto zero = [&](float* p, size_t n) {
    fillz_kernel<<<(unsigned)((n + 255) / 256), 256, 0, stream>>>(p, n);
  };

  const float* curIn = x;
  for (int L = 0; L < 2; ++L) {
    const int m = 4 * L;
    float* encOut = (L == 0) ? nodeA : nodeB;
    if (L == 0)
      mlp_fused<256, 128, 128, 128, true, true, 4><<<gN4, 128, 0, stream>>>(
          curIn, NN, P[m + 0], Wt0[m + 0], Wt1[m + 0], encOut);
    else
      mlp_fused<128, 128, 128, 128, true, true, 4><<<gN4, 128, 0, stream>>>(
          curIn, NN, P[m + 0], Wt0[m + 0], Wt1[m + 0], encOut);

    // V -> E : gather by src node, scatter to (he - min)
    zero(heA, (size_t)NHE * 128);
    scatter_kernel<<<sG, 256, 0, stream>>>(encOut, src, he, norm, minhe, 0, 1, heA, NNZ);
    mlp_fused<128, 128, 128, 128, true, true, 4><<<gE4, 128, 0, stream>>>(
        heA, NHE, P[m + 1], Wt0[m + 1], Wt1[m + 1], heB);          // v2e.dec
    mlp_fused<128, 128, 128, 128, true, true, 4><<<gE4, 128, 0, stream>>>(
        heB, NHE, P[m + 2], Wt0[m + 2], Wt1[m + 2], heA);          // e2v.enc

    // E -> V : gather by (he - min), scatter to src node
    zero(nodeB, (size_t)NN * 128);
    scatter_kernel<<<sG, 256, 0, stream>>>(heA, he, src, norm, minhe, 1, 0, nodeB, NNZ);
    mlp_fused<128, 128, 128, 128, true, true, 4><<<gN4, 128, 0, stream>>>(
        nodeB, NN, P[m + 3], Wt0[m + 3], Wt1[m + 3], nodeA);       // e2v.dec
    curIn = nodeA;
  }

  // classifier head: 128 -> 256 -> 40 (padded to 48 tiles), logits (no relu)
  mlp_fused<128, 256, 48, 40, false, false, 2><<<gN2, 128, 0, stream>>>(
      nodeA, NN, P[8], Wt0[8], Wt1[8], (float*)d_out);
}
